// ObjectNerf_22419729285322
// MI455X (gfx1250) — matrix operand order repaired and checked by reading, not compiled, not run on hardware
//
#include <hip/hip_runtime.h>
#include <hip/hip_bf16.h>
#include <stdint.h>

// ---------------------------------------------------------------------------
// Types for WMMA fragments (gfx1250, wave32)
// ---------------------------------------------------------------------------
typedef _Float16 v8h  __attribute__((ext_vector_type(8)));
typedef _Float16 v16h __attribute__((ext_vector_type(16)));
typedef float    v8f  __attribute__((ext_vector_type(8)));

#define NERF_B   1024
#define NERF_P   128
#define NERF_T   (NERF_B * NERF_P)   // 131072 samples
#define NERF_N   8
#define NERF_HID 128
#define NERF_KP  96                   // IN_DIM=66 padded to 3*32
#define PI_F     3.14159265358979323846f

// LDS row stride for 128-wide f16 activations: 136 f16 = 272B.
// 272 % 16 == 0 (aligned b128 reads) and 272/4 = 68 banks -> rows rotate by
// 4 banks, so the 16-row fragment reads are bank-conflict free.
#define HS 136

__device__ __forceinline__ v16h cat8h(v8h lo, v8h hi) {
    return __builtin_shufflevector(lo, hi, 0,1,2,3,4,5,6,7,8,9,10,11,12,13,14,15);
}

// ---------------------------------------------------------------------------
// Kernel 1: scalar_array + diffs. Reproduces jax.random.uniform(key(42),(128,))
// with a real Threefry-2x32 (JAX's default PRNG).
// ---------------------------------------------------------------------------
__device__ __forceinline__ uint32_t tf_rotl(uint32_t x, int r) {
    return (x << r) | (x >> (32 - r));
}
__device__ __forceinline__ void tf_round(uint32_t& x0, uint32_t& x1, int r) {
    x0 += x1; x1 = tf_rotl(x1, r); x1 ^= x0;
}
__device__ uint2 threefry2x32(uint32_t k0, uint32_t k1, uint32_t x0, uint32_t x1) {
    uint32_t k2 = k0 ^ k1 ^ 0x1BD11BDAu;
    x0 += k0; x1 += k1;
    tf_round(x0,x1,13); tf_round(x0,x1,15); tf_round(x0,x1,26); tf_round(x0,x1,6);
    x0 += k1; x1 += k2 + 1u;
    tf_round(x0,x1,17); tf_round(x0,x1,29); tf_round(x0,x1,16); tf_round(x0,x1,24);
    x0 += k2; x1 += k0 + 2u;
    tf_round(x0,x1,13); tf_round(x0,x1,15); tf_round(x0,x1,26); tf_round(x0,x1,6);
    x0 += k0; x1 += k1 + 3u;
    tf_round(x0,x1,17); tf_round(x0,x1,29); tf_round(x0,x1,16); tf_round(x0,x1,24);
    x0 += k1; x1 += k2 + 4u;
    tf_round(x0,x1,13); tf_round(x0,x1,15); tf_round(x0,x1,26); tf_round(x0,x1,6);
    x0 += k2; x1 += k0 + 5u;
    return make_uint2(x0, x1);
}

__global__ void prep_scalar_kernel(const float* __restrict__ s_start,
                                   const float* __restrict__ s_end,
                                   float* __restrict__ scalar_arr,
                                   float* __restrict__ diffs) {
    __shared__ float sh[NERF_P];
    int p = threadIdx.x;                 // 128 threads
    float start = *s_start, end = *s_end;
    // JAX: key(42) -> (0, 42); counts iota(128) split into halves.
    uint32_t bits;
    if (p < 64) bits = threefry2x32(0u, 42u, (uint32_t)p, (uint32_t)(p + 64)).x;
    else        bits = threefry2x32(0u, 42u, (uint32_t)(p - 64), (uint32_t)p).y;
    float u = __uint_as_float((bits >> 9) | 0x3F800000u) - 1.0f;   // [0,1)
    float step = (end - start) / (float)(NERF_P - 1);
    float sc = (start + step * (float)p) + u * step;
    sh[p] = sc;
    scalar_arr[p] = sc;
    __syncthreads();
    if (p < NERF_P - 1) diffs[p] = sh[p + 1] - sh[p];
}

// ---------------------------------------------------------------------------
// Kernel 2a/2b: weight transpose+convert to f16, layout [n][h][k] so a lane's
// WMMA B-fragment (16 contiguous K for its column) is one 32B read.
// ---------------------------------------------------------------------------
__global__ void prep_w1_kernel(const float* __restrict__ W1, _Float16* __restrict__ W1h) {
    int i = blockIdx.x * 256 + threadIdx.x;          // [8*128*96)
    if (i >= NERF_N * NERF_HID * NERF_KP) return;
    int k = i % NERF_KP;
    int h = (i / NERF_KP) % NERF_HID;
    int n = i / (NERF_KP * NERF_HID);
    float v = (k < 66) ? W1[(n * 66 + k) * NERF_HID + h] : 0.0f;   // W1[n][k][h]
    W1h[i] = (_Float16)v;
}

__global__ void prep_w2_kernel(const float* __restrict__ W2, _Float16* __restrict__ W2h) {
    int i = blockIdx.x * 256 + threadIdx.x;          // [8*128*128)
    if (i >= NERF_N * NERF_HID * NERF_HID) return;
    int k = i % NERF_HID;
    int h = (i / NERF_HID) % NERF_HID;
    int n = i / (NERF_HID * NERF_HID);
    W2h[i] = (_Float16)W2[(n * NERF_HID + k) * NERF_HID + h];      // W2[n][k][h]
}

// ---------------------------------------------------------------------------
// Kernel 3: positional encoding -> f16 X[t][96] (padded) + softmax splits.
// ---------------------------------------------------------------------------
__global__ __launch_bounds__(256)
void encode_kernel(const float* __restrict__ ray_o, const float* __restrict__ ray_d,
                   const float* __restrict__ scalar_arr,
                   const float* __restrict__ Ws, const float* __restrict__ bs,
                   _Float16* __restrict__ Xh, float* __restrict__ splits) {
    int t = blockIdx.x * 256 + threadIdx.x;
    if (t >= NERF_T) return;
    int b = t >> 7;
    int p = t & (NERF_P - 1);
    float s  = scalar_arr[p];
    float d0 = ray_d[b * 3 + 0], d1 = ray_d[b * 3 + 1], d2 = ray_d[b * 3 + 2];
    float r0 = ray_o[b * 3 + 0] + d0 * s;
    float r1 = ray_o[b * 3 + 1] + d1 * s;
    float r2 = ray_o[b * 3 + 2] + d2 * s;

    _Float16* x = Xh + (size_t)t * NERF_KP;
    x[0] = (_Float16)r0; x[1] = (_Float16)r1; x[2] = (_Float16)r2;
    float f = PI_F;                                   // 2^l * pi, l=0..9
    #pragma unroll
    for (int l = 0; l < 10; ++l) {
        x[3 + l * 6 + 0] = (_Float16)__cosf(f * r0);
        x[3 + l * 6 + 1] = (_Float16)__cosf(f * r1);
        x[3 + l * 6 + 2] = (_Float16)__cosf(f * r2);
        x[3 + l * 6 + 3] = (_Float16)__sinf(f * r0);
        x[3 + l * 6 + 4] = (_Float16)__sinf(f * r1);
        x[3 + l * 6 + 5] = (_Float16)__sinf(f * r2);
        f *= 2.0f;
    }
    x[63] = (_Float16)d0; x[64] = (_Float16)d1; x[65] = (_Float16)d2;
    #pragma unroll
    for (int k = 66; k < NERF_KP; ++k) x[k] = (_Float16)0.0f;

    // splits = softmax(ray @ Ws + bs) over N=8
    float lg[NERF_N];
    float mx = -1e30f;
    #pragma unroll
    for (int n = 0; n < NERF_N; ++n) {
        lg[n] = bs[n] + r0 * Ws[0 * NERF_N + n] + r1 * Ws[1 * NERF_N + n]
                      + r2 * Ws[2 * NERF_N + n];
        mx = fmaxf(mx, lg[n]);
    }
    float sum = 0.0f;
    #pragma unroll
    for (int n = 0; n < NERF_N; ++n) { lg[n] = __expf(lg[n] - mx); sum += lg[n]; }
    float inv = 1.0f / sum;
    #pragma unroll
    for (int n = 0; n < NERF_N; ++n) splits[(size_t)t * NERF_N + n] = lg[n] * inv;
}

// ---------------------------------------------------------------------------
// Kernel 4: fused 8-ensemble MLP on WMMA.
// Block = 256 threads (8 waves), 128-sample tile. Wave w owns output columns
// [16w,16w+16). 8 M-tile accumulators live in registers across the K loop.
// ---------------------------------------------------------------------------
__global__ __launch_bounds__(256)
void mlp_kernel(const _Float16* __restrict__ Xh,     // [T][96]
                const _Float16* __restrict__ W1h,    // [8][128][96]
                const _Float16* __restrict__ W2h,    // [8][128][128]
                const float* __restrict__ b1,        // [8][128]
                const float* __restrict__ b2,        // [8][128]
                const float* __restrict__ W3,        // [8][128][4]
                const float* __restrict__ b3,        // [8][4]
                const float* __restrict__ splits,    // [T][8]
                float* __restrict__ nerf_eval) {     // [T][4]
    __shared__ __align__(16) _Float16 H1[128 * HS];
    __shared__ __align__(16) _Float16 H2[128 * HS];

    const int tid  = threadIdx.x;
    const int wave = tid >> 5;
    const int lane = tid & 31;
    const int half = lane >> 4;          // which 16-lane half
    const int l15  = lane & 15;
    const int col  = wave * 16 + l15;    // output column this lane owns
    const int tileBase = blockIdx.x * 128;

    // layer-3 / blending assignment: thread pair per sample row
    const int orow = tid >> 1;           // 0..127
    const int cb   = (tid & 1) * 2;      // channels {0,1} or {2,3}
    float oacc0 = 0.0f, oacc1 = 0.0f;

    for (int n = 0; n < NERF_N; ++n) {
        // ---------------- Layer 1: X[128x96] @ W1[96x128] ----------------
        v8f acc[8];
        {
            float bias = b1[n * NERF_HID + col];
            #pragma unroll
            for (int mi = 0; mi < 8; ++mi)
                #pragma unroll
                for (int e = 0; e < 8; ++e) acc[mi][e] = bias;
        }
        #pragma unroll
        for (int ks = 0; ks < 3; ++ks) {
            const _Float16* bp =
                W1h + ((size_t)(n * NERF_HID + col) * NERF_KP + ks * 32 + 16 * half);
            v16h bf = cat8h(*(const v8h*)bp, *(const v8h*)(bp + 8));
            #pragma unroll
            for (int mi = 0; mi < 8; ++mi) {
                const _Float16* ap =
                    Xh + ((size_t)(tileBase + mi * 16 + l15) * NERF_KP + ks * 32 + 16 * half);
                v16h af = cat8h(*(const v8h*)ap, *(const v8h*)(ap + 8));
                acc[mi] = __builtin_amdgcn_wmma_f32_16x16x32_f16(
                    false, af, false, bf, (short)0, acc[mi], false, false);
            }
        }
        #pragma unroll
        for (int mi = 0; mi < 8; ++mi)
            #pragma unroll
            for (int r = 0; r < 8; ++r) {
                int row = mi * 16 + r + 8 * half;
                H1[row * HS + col] = (_Float16)fmaxf(acc[mi][r], 0.0f);
            }
        __syncthreads();

        // ---------------- Layer 2: H1[128x128] @ W2[128x128] ----------------
        {
            float bias = b2[n * NERF_HID + col];
            #pragma unroll
            for (int mi = 0; mi < 8; ++mi)
                #pragma unroll
                for (int e = 0; e < 8; ++e) acc[mi][e] = bias;
        }
        #pragma unroll
        for (int ks = 0; ks < 4; ++ks) {
            const _Float16* bp =
                W2h + ((size_t)(n * NERF_HID + col) * NERF_HID + ks * 32 + 16 * half);
            v16h bf = cat8h(*(const v8h*)bp, *(const v8h*)(bp + 8));
            #pragma unroll
            for (int mi = 0; mi < 8; ++mi) {
                const _Float16* ap =
                    &H1[(mi * 16 + l15) * HS + ks * 32 + 16 * half];
                v16h af = cat8h(*(const v8h*)ap, *(const v8h*)(ap + 8));
                acc[mi] = __builtin_amdgcn_wmma_f32_16x16x32_f16(
                    false, af, false, bf, (short)0, acc[mi], false, false);
            }
        }
        #pragma unroll
        for (int mi = 0; mi < 8; ++mi)
            #pragma unroll
            for (int r = 0; r < 8; ++r) {
                int row = mi * 16 + r + 8 * half;
                H2[row * HS + col] = (_Float16)fmaxf(acc[mi][r], 0.0f);
            }
        __syncthreads();

        // ---------------- Layer 3: H2[128x128] @ W3[128x4] + blend ----------
        {
            float e0 = b3[n * 4 + cb];
            float e1 = b3[n * 4 + cb + 1];
            #pragma unroll 8
            for (int k = 0; k < NERF_HID; ++k) {
                float h = (float)H2[orow * HS + k];
                float2 w = *(const float2*)&W3[(size_t)(n * NERF_HID + k) * 4 + cb];
                e0 = fmaf(h, w.x, e0);
                e1 = fmaf(h, w.y, e1);
            }
            float sp = splits[(size_t)(tileBase + orow) * NERF_N + n];
            oacc0 = fmaf(e0, sp, oacc0);
            oacc1 = fmaf(e1, sp, oacc1);
        }
        __syncthreads();   // protect H1/H2 before next ensemble overwrites
    }

    float2 o = make_float2(oacc0, oacc1);
    *(float2*)&nerf_eval[(size_t)(tileBase + orow) * 4 + cb] = o;
}

// ---------------------------------------------------------------------------
// Kernel 5: volume rendering, one thread per ray (127-step scan).
// ---------------------------------------------------------------------------
__global__ void render_kernel(const float* __restrict__ nerf_eval,
                              const float* __restrict__ diffs,
                              float* __restrict__ rendered) {
    int b = blockIdx.x * 256 + threadIdx.x;
    if (b >= NERF_B) return;
    float a0 = 0.0f, a1 = 0.0f, a2 = 0.0f, cum = 0.0f;
    for (int p = 0; p < NERF_P - 1; ++p) {
        float4 e = *(const float4*)&nerf_eval[(size_t)(b * NERF_P + p) * 4];
        float sd = e.w * diffs[p];
        cum += sd;
        float T   = __expf(cum);         // reference: T = exp(cumsum(sd))
        float mid = 1.0f - __expf(-sd);
        float w = T * mid;
        a0 = fmaf(w, e.x, a0);
        a1 = fmaf(w, e.y, a1);
        a2 = fmaf(w, e.z, a2);
    }
    rendered[b * 3 + 0] = a0;
    rendered[b * 3 + 1] = a1;
    rendered[b * 3 + 2] = a2;
}

// ---------------------------------------------------------------------------
// Host launcher
// ---------------------------------------------------------------------------
extern "C" void kernel_launch(void* const* d_in, const int* in_sizes, int n_in,
                              void* d_out, int out_size, void* d_ws, size_t ws_size,
                              hipStream_t stream) {
    const float* ray_o   = (const float*)d_in[0];
    const float* ray_d   = (const float*)d_in[1];
    const float* s_start = (const float*)d_in[2];
    const float* s_end   = (const float*)d_in[3];
    // d_in[4] = num_integration_points (==128, hardcoded)
    const float* W1 = (const float*)d_in[5];
    const float* b1 = (const float*)d_in[6];
    const float* W2 = (const float*)d_in[7];
    const float* b2 = (const float*)d_in[8];
    const float* W3 = (const float*)d_in[9];
    const float* b3 = (const float*)d_in[10];
    const float* Ws = (const float*)d_in[11];
    const float* bs = (const float*)d_in[12];

    float* out      = (float*)d_out;
    float* rendered = out;                       // [1024*3]
    float* splits   = out + NERF_B * 3;          // [131072*8]

    char* ws = (char*)d_ws;
    size_t off = 0;
    float* scalar_arr = (float*)(ws + off);  off += 512;                       // 128 f32
    float* diffs      = (float*)(ws + off);  off += 512;                       // 127 f32
    _Float16* W1h = (_Float16*)(ws + off);   off += (size_t)NERF_N*NERF_HID*NERF_KP*2;
    _Float16* W2h = (_Float16*)(ws + off);   off += (size_t)NERF_N*NERF_HID*NERF_HID*2;
    _Float16* Xh  = (_Float16*)(ws + off);   off += (size_t)NERF_T*NERF_KP*2;
    float* neval  = (float*)(ws + off);      off += (size_t)NERF_T*4*4;
    (void)ws_size; (void)in_sizes; (void)n_in; (void)out_size;

    prep_scalar_kernel<<<1, 128, 0, stream>>>(s_start, s_end, scalar_arr, diffs);
    prep_w1_kernel<<<(NERF_N*NERF_HID*NERF_KP + 255)/256, 256, 0, stream>>>(W1, W1h);
    prep_w2_kernel<<<(NERF_N*NERF_HID*NERF_HID + 255)/256, 256, 0, stream>>>(W2, W2h);
    encode_kernel<<<NERF_T/256, 256, 0, stream>>>(ray_o, ray_d, scalar_arr, Ws, bs, Xh, splits);
    mlp_kernel<<<NERF_T/128, 256, 0, stream>>>(Xh, W1h, W2h, b1, b2, W3, b3, splits, neval);
    render_kernel<<<(NERF_B + 255)/256, 256, 0, stream>>>(neval, diffs, rendered);
}
